// GCN_77120432767675
// MI455X (gfx1250) — compile-verified, matrix operand order checked by reading, not run on hardware
//
#include <hip/hip_runtime.h>
#include <hip/hip_bf16.h>
#include <math.h>

// ---------------- CDNA5 WMMA types ----------------
typedef __attribute__((ext_vector_type(16))) __bf16 v16bf;
typedef __attribute__((ext_vector_type(8)))  float  v8f;

#define NFEAT  512
#define NHID   256
#define NCLASS 64

__device__ __forceinline__ unsigned short f2bf_u(float f) {
  // round-to-nearest-even f32 -> bf16 (bit pattern)
  unsigned u = __float_as_uint(f);
  unsigned r = u + 0x7FFFu + ((u >> 16) & 1u);
  return (unsigned short)(r >> 16);
}

union Frag16 {
  v16bf v;
  unsigned short u[16];
  uint4 q[2];
};

// ---------------- weight convert + transpose (f32 [K,N] -> bf16 [N,K]) ----------------
__global__ __launch_bounds__(256) void cvtw_kernel(const float* __restrict__ W,
                                                   unsigned short* __restrict__ Wt,
                                                   int K, int N) {
  int i = blockIdx.x * 256 + threadIdx.x;
  if (i >= K * N) return;
  int n = i / K, k = i % K;
  Wt[i] = f2bf_u(W[(size_t)k * N + n]);
}

// ---------------- GEMM1: S0[M,256] = x[M,512](f32) @ W1 (bf16 WMMA) ----------------
// block = 256 thr = 8 waves: 2 row-strips x 4 col-groups; wave: 16 rows x 64 cols (4 tiles)
__global__ __launch_bounds__(256) void gemm1_kernel(const float* __restrict__ x,
                                                    const unsigned short* __restrict__ W1t,
                                                    float* __restrict__ S0, int M) {
  int tid   = threadIdx.x;
  int lane  = tid & 31;
  int wave  = tid >> 5;
  int strip = wave >> 2;          // 0..1
  int cg    = wave & 3;           // col group: 64 cols each
  int rowbase = blockIdx.x * 32 + strip * 16;
  int l16   = lane & 15;
  int lhalf = lane >> 4;          // lane half selects K sub-block per ISA layout
  int kb    = lhalf * 8;
  int row   = rowbase + l16;
  int rowc  = row < M ? row : (M - 1);

  v8f acc[4];
#pragma unroll
  for (int t = 0; t < 4; ++t)
#pragma unroll
    for (int v = 0; v < 8; ++v) acc[t][v] = 0.f;

  for (int k0 = 0; k0 < NFEAT; k0 += 32) {
    // A fragment: row 'row', K = k0+kb..+7 and k0+kb+16..+23 (f32 -> bf16)
    Frag16 a;
    const float* ap = x + (size_t)rowc * NFEAT + k0 + kb;
    float4 f0 = *(const float4*)(ap + 0);
    float4 f1 = *(const float4*)(ap + 4);
    float4 f2 = *(const float4*)(ap + 16);
    float4 f3 = *(const float4*)(ap + 20);
    a.u[0] = f2bf_u(f0.x);  a.u[1] = f2bf_u(f0.y);
    a.u[2] = f2bf_u(f0.z);  a.u[3] = f2bf_u(f0.w);
    a.u[4] = f2bf_u(f1.x);  a.u[5] = f2bf_u(f1.y);
    a.u[6] = f2bf_u(f1.z);  a.u[7] = f2bf_u(f1.w);
    a.u[8]  = f2bf_u(f2.x); a.u[9]  = f2bf_u(f2.y);
    a.u[10] = f2bf_u(f2.z); a.u[11] = f2bf_u(f2.w);
    a.u[12] = f2bf_u(f3.x); a.u[13] = f2bf_u(f3.y);
    a.u[14] = f2bf_u(f3.z); a.u[15] = f2bf_u(f3.w);

#pragma unroll
    for (int t = 0; t < 4; ++t) {
      int n = cg * 64 + t * 16 + l16;       // B column owned by this lane
      Frag16 b;
      const uint4* bp = (const uint4*)(W1t + (size_t)n * NFEAT + k0 + kb);
      b.q[0] = bp[0];                        // K = k0+kb   .. +7
      b.q[1] = bp[2];                        // K = k0+kb+16.. +23
      acc[t] = __builtin_amdgcn_wmma_f32_16x16x32_bf16(
          false, a.v, false, b.v, (short)0, acc[t], false, false);
    }
  }

  // C/D layout: VGPR v, lanes 0-15 -> row v, lanes 16-31 -> row v+8; col = lane%16
  int r0 = rowbase + lhalf * 8;
  if (rowbase + 16 <= M) {
    // full tile: straight-line stores, no exec manipulation
#pragma unroll
    for (int t = 0; t < 4; ++t) {
      int col = cg * 64 + t * 16 + l16;
      float* op = S0 + (size_t)r0 * NHID + col;
#pragma unroll
      for (int v = 0; v < 8; ++v) op[(size_t)v * NHID] = acc[t][v];
    }
  } else {
#pragma unroll
    for (int t = 0; t < 4; ++t) {
      int col = cg * 64 + t * 16 + l16;
#pragma unroll
      for (int v = 0; v < 8; ++v) {
        int r = r0 + v;
        if (r < M) S0[(size_t)r * NHID + col] = acc[t][v];
      }
    }
  }
}

// ---------------- GEMM2: S1[M,64] = H0[M,256](bf16) @ W2 (bf16 WMMA) ----------------
// block = 8 waves = 8 row-strips x 16 rows; wave covers all 64 cols (4 tiles)
__global__ __launch_bounds__(256) void gemm2_kernel(const unsigned short* __restrict__ H0,
                                                    const unsigned short* __restrict__ W2t,
                                                    float* __restrict__ S1, int M) {
  int tid   = threadIdx.x;
  int lane  = tid & 31;
  int wave  = tid >> 5;
  int rowbase = blockIdx.x * 128 + wave * 16;
  int l16   = lane & 15;
  int lhalf = lane >> 4;
  int kb    = lhalf * 8;
  int row   = rowbase + l16;
  int rowc  = row < M ? row : (M - 1);

  v8f acc[4];
#pragma unroll
  for (int t = 0; t < 4; ++t)
#pragma unroll
    for (int v = 0; v < 8; ++v) acc[t][v] = 0.f;

  for (int k0 = 0; k0 < NHID; k0 += 32) {
    Frag16 a;
    const uint4* ap = (const uint4*)(H0 + (size_t)rowc * NHID + k0 + kb);
    a.q[0] = ap[0];
    a.q[1] = ap[2];
#pragma unroll
    for (int t = 0; t < 4; ++t) {
      int n = t * 16 + l16;
      Frag16 b;
      const uint4* bp = (const uint4*)(W2t + (size_t)n * NHID + k0 + kb);
      b.q[0] = bp[0];
      b.q[1] = bp[2];
      acc[t] = __builtin_amdgcn_wmma_f32_16x16x32_bf16(
          false, a.v, false, b.v, (short)0, acc[t], false, false);
    }
  }

  int r0 = rowbase + lhalf * 8;
  if (rowbase + 16 <= M) {
#pragma unroll
    for (int t = 0; t < 4; ++t) {
      int col = t * 16 + l16;
      float* op = S1 + (size_t)r0 * NCLASS + col;
#pragma unroll
      for (int v = 0; v < 8; ++v) op[(size_t)v * NCLASS] = acc[t][v];
    }
  } else {
#pragma unroll
    for (int t = 0; t < 4; ++t) {
      int col = t * 16 + l16;
#pragma unroll
      for (int v = 0; v < 8; ++v) {
        int r = r0 + v;
        if (r < M) S1[(size_t)r * NCLASS + col] = acc[t][v];
      }
    }
  }
}

// ---------------- Z init: broadcast bias (so SpMM accumulates on top) ----------------
__global__ __launch_bounds__(256) void initz_kernel(float4* __restrict__ Z,
                                                    const float* __restrict__ b,
                                                    int nvec, int dmask) {
  int i = blockIdx.x * 256 + threadIdx.x;
  if (i >= nvec) return;
  int f = (i * 4) & dmask;
  Z[i] = make_float4(b[f], b[f + 1], b[f + 2], b[f + 3]);
}

// ---------------- SpMM scatter: Z[dst] += w * S[src] ----------------
// 64 threads/edge (d=256), 4 contiguous feats per thread, f32 global atomics (L2-resident)
__global__ __launch_bounds__(256) void spmm1_kernel(const int* __restrict__ src,
                                                    const int* __restrict__ dst,
                                                    const float* __restrict__ w,
                                                    const float* __restrict__ S,
                                                    float* __restrict__ Z, int E) {
  long long tid = (long long)blockIdx.x * 256 + threadIdx.x;
  int e = (int)(tid >> 6);
  if (e >= E) return;
  int f = ((int)tid & 63) * 4;
  int s = src[e], d = dst[e];
  float ww = w[e];
  float4 v = *(const float4*)(S + (size_t)s * NHID + f);
  float* zp = Z + (size_t)d * NHID + f;
  atomicAdd(zp + 0, ww * v.x);
  atomicAdd(zp + 1, ww * v.y);
  atomicAdd(zp + 2, ww * v.z);
  atomicAdd(zp + 3, ww * v.w);
}

// 16 threads/edge (d=64)
__global__ __launch_bounds__(256) void spmm2_kernel(const int* __restrict__ src,
                                                    const int* __restrict__ dst,
                                                    const float* __restrict__ w,
                                                    const float* __restrict__ S,
                                                    float* __restrict__ Z, int E) {
  long long tid = (long long)blockIdx.x * 256 + threadIdx.x;
  int e = (int)(tid >> 4);
  if (e >= E) return;
  int f = ((int)tid & 15) * 4;
  int s = src[e], d = dst[e];
  float ww = w[e];
  float4 v = *(const float4*)(S + (size_t)s * NCLASS + f);
  float* zp = Z + (size_t)d * NCLASS + f;
  atomicAdd(zp + 0, ww * v.x);
  atomicAdd(zp + 1, ww * v.y);
  atomicAdd(zp + 2, ww * v.z);
  atomicAdd(zp + 3, ww * v.w);
}

// ---------------- relu + f32->bf16 (H0 for GEMM2) ----------------
__global__ __launch_bounds__(256) void relu_bf16_kernel(const float4* __restrict__ Z,
                                                        ushort4* __restrict__ H, int nvec) {
  int i = blockIdx.x * 256 + threadIdx.x;
  if (i >= nvec) return;
  float4 z = Z[i];
  ushort4 h;
  h.x = f2bf_u(fmaxf(z.x, 0.f));
  h.y = f2bf_u(fmaxf(z.y, 0.f));
  h.z = f2bf_u(fmaxf(z.z, 0.f));
  h.w = f2bf_u(fmaxf(z.w, 0.f));
  H[i] = h;
}

// ---------------- log_softmax over 64 classes: one wave32 per row ----------------
__global__ __launch_bounds__(256) void logsoftmax_kernel(const float* __restrict__ Z,
                                                         float* __restrict__ out, int M) {
  int wave = (int)((blockIdx.x * 256 + threadIdx.x) >> 5);
  int lane = threadIdx.x & 31;
  if (wave >= M) return;
  const float* zp = Z + (size_t)wave * NCLASS;
  float a = zp[lane];
  float b = zp[lane + 32];
  float m = fmaxf(a, b);
#pragma unroll
  for (int o = 16; o > 0; o >>= 1) m = fmaxf(m, __shfl_xor(m, o, 32));
  float s = __expf(a - m) + __expf(b - m);
#pragma unroll
  for (int o = 16; o > 0; o >>= 1) s += __shfl_xor(s, o, 32);
  float ls = __logf(s);
  float* op = out + (size_t)wave * NCLASS;
  op[lane]      = a - m - ls;
  op[lane + 32] = b - m - ls;
}

extern "C" void kernel_launch(void* const* d_in, const int* in_sizes, int n_in,
                              void* d_out, int out_size, void* d_ws, size_t ws_size,
                              hipStream_t stream) {
  const float* x   = (const float*)d_in[0];
  const int*  esrc = (const int*)d_in[1];
  const int*  edst = (const int*)d_in[2];
  const float* ew  = (const float*)d_in[3];
  const float* W1  = (const float*)d_in[4];
  const float* b1  = (const float*)d_in[5];
  const float* W2  = (const float*)d_in[6];
  const float* b2  = (const float*)d_in[7];
  float* out = (float*)d_out;

  int M = in_sizes[0] / NFEAT;   // 100000 nodes
  int E = in_sizes[1];           // 3.2M edges

  // workspace layout (~205 MB):
  //   [W1t bf16 256KB][W2t bf16 32KB][S0 f32 / H0 bf16 102.4MB][Z0 f32 / (S1+Z1) 102.4MB]
  char* ws = (char*)d_ws;
  unsigned short* W1t = (unsigned short*)(ws);
  unsigned short* W2t = (unsigned short*)(ws + 262144);
  size_t oS0 = 262144 + 32768;
  float*          S0  = (float*)(ws + oS0);
  unsigned short* H0  = (unsigned short*)(ws + oS0);                 // reuse S0 region
  size_t oZ0 = oS0 + (size_t)M * NHID * sizeof(float);
  float* Z0 = (float*)(ws + oZ0);
  float* S1 = (float*)(ws + oZ0);                                    // reuse Z0 region
  float* Z1 = (float*)(ws + oZ0 + (size_t)M * NCLASS * sizeof(float));

  // weights -> transposed bf16
  cvtw_kernel<<<(NHID * NFEAT + 255) / 256, 256, 0, stream>>>(W1, W1t, NFEAT, NHID);
  cvtw_kernel<<<(NCLASS * NHID + 255) / 256, 256, 0, stream>>>(W2, W2t, NHID, NCLASS);

  // layer 1
  gemm1_kernel<<<(M + 31) / 32, 256, 0, stream>>>(x, W1t, S0, M);
  int nv0 = M * NHID / 4;
  initz_kernel<<<(nv0 + 255) / 256, 256, 0, stream>>>((float4*)Z0, b1, nv0, NHID - 1);
  int sb1 = (int)(((long long)E * 64 + 255) / 256);
  spmm1_kernel<<<sb1, 256, 0, stream>>>(esrc, edst, ew, S0, Z0, E);
  relu_bf16_kernel<<<(nv0 + 255) / 256, 256, 0, stream>>>((const float4*)Z0, (ushort4*)H0, nv0);

  // layer 2
  gemm2_kernel<<<(M + 127) / 128, 256, 0, stream>>>(H0, W2t, S1, M);
  int nv1 = M * NCLASS / 4;
  initz_kernel<<<(nv1 + 255) / 256, 256, 0, stream>>>((float4*)Z1, b2, nv1, NCLASS - 1);
  int sb2 = (int)(((long long)E * 16 + 255) / 256);
  spmm2_kernel<<<sb2, 256, 0, stream>>>(esrc, edst, ew, S1, Z1, E);

  logsoftmax_kernel<<<(M + 7) / 8, 256, 0, stream>>>(Z1, out, M);
}
